// RepairNet_53145925321043
// MI455X (gfx1250) — compile-verified, matrix operand order checked by reading, not run on hardware
//
#include <hip/hip_runtime.h>

typedef _Float16 v16h __attribute__((ext_vector_type(16)));
typedef _Float16 v8h  __attribute__((ext_vector_type(8)));
typedef float    v8f  __attribute__((ext_vector_type(8)));

#define NN 60
#define NE_BASE 1800
#define NE 1860
#define KP 416     // padded K for fc1 (400 real + 16 zeros)
#define NOUT 512

// ---------------------------------------------------------------------------
// Kernel 1: entire GNN branch (2x GAT + batchnorm + relu) + row-constant part
// of fc1: gbias[j] = b1[j] + sum_i g[i] * W1[i, j]  for i in [0,960)
// Single block, 256 threads, fully deterministic (no float atomics).
// ---------------------------------------------------------------------------
__device__ void gat_layer(int tid, int in_dim, int out_dim,
    const float* __restrict__ W, const float* __restrict__ a_src,
    const float* __restrict__ a_dst, const float* __restrict__ lin_e,
    const float* __restrict__ a_e, const float* __restrict__ bias,
    const float* __restrict__ edge_fea, float eamean,
    float* s_x, float* s_xl, float* s_asrc, float* s_adst,
    float* s_alpha, float* s_maxf, float* s_denom,
    const int* s_src, const int* s_dst)
{
    // xl = x @ W ; per-node attention logits
    for (int i = tid; i < NN; i += 256) {
        float as = 0.f, ad = 0.f;
        for (int k = 0; k < out_dim; ++k) {
            float v = 0.f;
            for (int j = 0; j < in_dim; ++j) v += s_x[i*16 + j] * W[j*out_dim + k];
            s_xl[i*16 + k] = v;
            as += v * a_src[k];
            ad += v * a_dst[k];
        }
        s_asrc[i] = as; s_adst[i] = ad;
    }
    float ecoef = 0.f;
    for (int k = 0; k < out_dim; ++k) ecoef += lin_e[k] * a_e[k];
    __syncthreads();

    // per-edge raw attention (leaky relu)
    for (int e = tid; e < NE; e += 256) {
        float ea = (e < NE_BASE) ? edge_fea[e] : eamean;
        float v = s_asrc[s_src[e]] + s_adst[s_dst[e]] + ea * ecoef;
        s_alpha[e] = (v > 0.f) ? v : 0.2f * v;
    }
    __syncthreads();

    // segment max + denom per dst node (deterministic scan)
    for (int i = tid; i < NN; i += 256) {
        float m = -3.4e38f;
        for (int e = 0; e < NE; ++e)
            if (s_dst[e] == i) { float v = s_alpha[e]; m = v > m ? v : m; }
        float den = 0.f;
        for (int e = 0; e < NE; ++e)
            if (s_dst[e] == i) den += __expf(s_alpha[e] - m);
        s_maxf[i] = m; s_denom[i] = den;
    }
    __syncthreads();

    // normalized edge weights
    for (int e = tid; e < NE; e += 256) {
        int d = s_dst[e];
        s_alpha[e] = __expf(s_alpha[e] - s_maxf[d]) / s_denom[d];
    }
    __syncthreads();

    // weighted aggregation: out[i,k] = sum_{e: dst==i} w_e * xl[src,k]
    for (int w = tid; w < NN * out_dim; w += 256) {
        int i = w / out_dim, k = w % out_dim;
        float acc = 0.f;
        for (int e = 0; e < NE; ++e)
            if (s_dst[e] == i) acc += s_alpha[e] * s_xl[s_src[e]*16 + k];
        s_x[i*16 + k] = acc + bias[k];
    }
    __syncthreads();
}

__device__ void bn_relu(int tid, int dim, float* s_x, float* s_red, float* s_stat)
{
    int total = NN * dim;
    float ps = 0.f;
    for (int w = tid; w < total; w += 256) ps += s_x[(w/dim)*16 + (w%dim)];
    s_red[tid] = ps;
    __syncthreads();
    if (tid == 0) { float s = 0.f; for (int i = 0; i < 256; ++i) s += s_red[i]; s_stat[0] = s / total; }
    __syncthreads();
    float mu = s_stat[0];
    ps = 0.f;
    for (int w = tid; w < total; w += 256) { float d = s_x[(w/dim)*16 + (w%dim)] - mu; ps += d * d; }
    s_red[tid] = ps;
    __syncthreads();
    if (tid == 0) { float s = 0.f; for (int i = 0; i < 256; ++i) s += s_red[i]; s_stat[1] = sqrtf(s / total); }
    __syncthreads();
    float inv = 1.f / s_stat[1];
    for (int w = tid; w < total; w += 256) {
        int idx = (w/dim)*16 + (w%dim);
        float v = (s_x[idx] - mu) * inv;
        s_x[idx] = v > 0.f ? v : 0.f;
    }
    __syncthreads();
}

__global__ __launch_bounds__(256) void gnn_kernel(
    const float* __restrict__ node_fea, const float* __restrict__ edge_fea,
    const int* __restrict__ edge_index,
    const float* g1_w, const float* g1_as, const float* g1_ad,
    const float* g1_le, const float* g1_ae, const float* g1_b,
    const float* g2_w, const float* g2_as, const float* g2_ad,
    const float* g2_le, const float* g2_ae, const float* g2_b,
    const float* __restrict__ fc1_w, const float* __restrict__ fc1_b,
    float* __restrict__ gbias)
{
    __shared__ float s_x[NN*16], s_xl[NN*16];
    __shared__ float s_asrc[NN], s_adst[NN];
    __shared__ float s_alpha[NE];
    __shared__ float s_maxf[NN], s_denom[NN];
    __shared__ int   s_src[NE], s_dst[NE];
    __shared__ float s_red[256];
    __shared__ float s_stat[2];
    int tid = threadIdx.x;

    // edges with self loops appended
    for (int e = tid; e < NE; e += 256) {
        if (e < NE_BASE) { s_src[e] = edge_index[e]; s_dst[e] = edge_index[NE_BASE + e]; }
        else             { s_src[e] = e - NE_BASE;   s_dst[e] = e - NE_BASE; }
    }
    // mean edge feature (self-loop fill value)
    float ps = 0.f;
    for (int e = tid; e < NE_BASE; e += 256) ps += edge_fea[e];
    s_red[tid] = ps;
    // initial node features (in_dim = 1)
    for (int i = tid; i < NN; i += 256) s_x[i*16] = node_fea[i];
    __syncthreads();
    if (tid == 0) { float s = 0.f; for (int i = 0; i < 256; ++i) s += s_red[i]; s_stat[0] = s / NE_BASE; }
    __syncthreads();
    float eamean = s_stat[0];

    gat_layer(tid, 1, 6,  g1_w, g1_as, g1_ad, g1_le, g1_ae, g1_b, edge_fea, eamean,
              s_x, s_xl, s_asrc, s_adst, s_alpha, s_maxf, s_denom, s_src, s_dst);
    bn_relu(tid, 6, s_x, s_red, s_stat);
    gat_layer(tid, 6, 16, g2_w, g2_as, g2_ad, g2_le, g2_ae, g2_b, edge_fea, eamean,
              s_x, s_xl, s_asrc, s_adst, s_alpha, s_maxf, s_denom, s_src, s_dst);
    bn_relu(tid, 16, s_x, s_red, s_stat);

    // g is s_x[0..960) contiguous (stride 16, dim 16). gbias = b1 + g @ W1[0:960]
    for (int j = tid; j < NOUT; j += 256) {
        float acc = fc1_b[j];
        for (int i = 0; i < 960; ++i) acc += s_x[i] * fc1_w[i*NOUT + j];
        gbias[j] = acc;
    }
}

// ---------------------------------------------------------------------------
// Kernel 2: conv1(3->6,5x5) + relu + maxpool2 fused. One thread per pooled out.
// ---------------------------------------------------------------------------
__global__ __launch_bounds__(256) void conv1_kernel(
    const float* __restrict__ pic, const float* __restrict__ w,
    const float* __restrict__ b, float* __restrict__ p1)
{
    __shared__ float s_w[450];
    __shared__ float s_b[6];
    for (int i = threadIdx.x; i < 450; i += 256) s_w[i] = w[i];
    if (threadIdx.x < 6) s_b[threadIdx.x] = b[threadIdx.x];
    __syncthreads();
    int idx = blockIdx.x * 256 + threadIdx.x;
    if (idx >= 8192 * 1176) return;
    int x = idx % 14, y = (idx / 14) % 14, c = (idx / 196) % 6, r = idx / 1176;
    float a00 = s_b[c], a01 = s_b[c], a10 = s_b[c], a11 = s_b[c];
    int iy0 = 2*y, ix0 = 2*x;
    for (int ic = 0; ic < 3; ++ic) {
        float patch[6][6];
        const float* pbase = pic + (((size_t)r*3 + ic)*32 + iy0)*32 + ix0;
        #pragma unroll
        for (int py = 0; py < 6; ++py)
            #pragma unroll
            for (int px = 0; px < 6; ++px) patch[py][px] = pbase[py*32 + px];
        const float* wc = s_w + (c*3 + ic)*25;
        #pragma unroll
        for (int ky = 0; ky < 5; ++ky)
            #pragma unroll
            for (int kx = 0; kx < 5; ++kx) {
                float wv = wc[ky*5 + kx];
                a00 += patch[ky  ][kx  ] * wv;
                a01 += patch[ky  ][kx+1] * wv;
                a10 += patch[ky+1][kx  ] * wv;
                a11 += patch[ky+1][kx+1] * wv;
            }
    }
    float m = fmaxf(fmaxf(a00, a01), fmaxf(a10, a11));
    p1[idx] = fmaxf(m, 0.f);   // relu(max) == max(relu)
}

// ---------------------------------------------------------------------------
// Kernel 3: conv2(6->16,5x5) + relu + maxpool2 fused; writes the fc1 A-matrix
// directly as f16 [8192, 416] (cols 400..415 zero padding).
// ---------------------------------------------------------------------------
__global__ __launch_bounds__(256) void conv2_kernel(
    const float* __restrict__ p1, const float* __restrict__ w,
    const float* __restrict__ b, _Float16* __restrict__ A)
{
    __shared__ float s_w[2400];
    __shared__ float s_b[16];
    for (int i = threadIdx.x; i < 2400; i += 256) s_w[i] = w[i];
    if (threadIdx.x < 16) s_b[threadIdx.x] = b[threadIdx.x];
    __syncthreads();
    int idx = blockIdx.x * 256 + threadIdx.x;
    if (idx >= 8192 * KP) return;
    int col = idx % KP;
    int r   = idx / KP;
    if (col >= 400) { A[idx] = (_Float16)0.f; return; }
    int c = col / 25, rem = col % 25, y = rem / 5, x = rem % 5;
    float a00 = s_b[c], a01 = s_b[c], a10 = s_b[c], a11 = s_b[c];
    int iy0 = 2*y, ix0 = 2*x;
    for (int ic = 0; ic < 6; ++ic) {
        float patch[6][6];
        const float* pbase = p1 + (((size_t)r*6 + ic)*14 + iy0)*14 + ix0;
        #pragma unroll
        for (int py = 0; py < 6; ++py)
            #pragma unroll
            for (int px = 0; px < 6; ++px) patch[py][px] = pbase[py*14 + px];
        const float* wc = s_w + (c*6 + ic)*25;
        #pragma unroll
        for (int ky = 0; ky < 5; ++ky)
            #pragma unroll
            for (int kx = 0; kx < 5; ++kx) {
                float wv = wc[ky*5 + kx];
                a00 += patch[ky  ][kx  ] * wv;
                a01 += patch[ky  ][kx+1] * wv;
                a10 += patch[ky+1][kx  ] * wv;
                a11 += patch[ky+1][kx+1] * wv;
            }
    }
    float m = fmaxf(fmaxf(a00, a01), fmaxf(a10, a11));
    A[idx] = (_Float16)fmaxf(m, 0.f);
}

// ---------------------------------------------------------------------------
// Kernel 4: transpose/pad fc1 weights rows 960..1359 into Bt [512][416] f16
// ---------------------------------------------------------------------------
__global__ __launch_bounds__(256) void prepb_kernel(
    const float* __restrict__ fc1_w, _Float16* __restrict__ Bt)
{
    int idx = blockIdx.x * 256 + threadIdx.x;
    if (idx >= NOUT * KP) return;
    int n = idx / KP, k = idx % KP;
    Bt[idx] = (k < 400) ? (_Float16)fc1_w[(960 + k)*NOUT + n] : (_Float16)0.f;
}

// ---------------------------------------------------------------------------
// Kernel 5: fc1 GEMM via v_wmma_f32_16x16x32_f16.
// C[8192,512] = A[8192,416] x Bt^T[416,512], + gbias[n], relu, f16 store.
// One wave -> 64x16 output strip (4 accumulators, B fragment reused 4x).
// Fragment layout (16-bit A, 16x32): lane l (l16=l&15, hi=l>>4) holds row
// M=l16; halves 0..7 -> K = k0+8*hi+h, halves 8..15 -> K = k0+16+8*hi+(h-8):
// two contiguous 16-byte global loads. Bt rows give the symmetric B layout.
// ---------------------------------------------------------------------------
union Frag { v16h v; v8h h[2]; };

__global__ __launch_bounds__(256) void fc1_wmma_kernel(
    const _Float16* __restrict__ A, const _Float16* __restrict__ Bt,
    const float* __restrict__ gbias, _Float16* __restrict__ H)
{
    int wave = blockIdx.x * 8 + (threadIdx.x >> 5);
    int lane = threadIdx.x & 31;
    int l16 = lane & 15, hi = lane >> 4;
    int ntile = wave & 31;        // 512/16 = 32 n-tiles
    int mtile = wave >> 5;        // 8192/64 = 128 m-strips
    int n0 = ntile * 16;
    int m0 = mtile * 64;

    v8f acc[4] = {v8f{}, v8f{}, v8f{}, v8f{}};
    const _Float16* brow = Bt + (size_t)(n0 + l16) * KP + 8*hi;
    const _Float16* arow = A  + (size_t)(m0 + l16) * KP + 8*hi;

    for (int ks = 0; ks < 13; ++ks) {
        int k0 = ks * 32;
        Frag bf;
        bf.h[0] = *(const v8h*)(brow + k0);
        bf.h[1] = *(const v8h*)(brow + k0 + 16);
        #pragma unroll
        for (int sub = 0; sub < 4; ++sub) {
            const _Float16* ap = arow + (size_t)sub * 16 * KP + k0;
            Frag af;
            af.h[0] = *(const v8h*)(ap);
            af.h[1] = *(const v8h*)(ap + 16);
            acc[sub] = __builtin_amdgcn_wmma_f32_16x16x32_f16(
                false, af.v, false, bf.v, (short)0, acc[sub], false, false);
        }
    }

    float bias = gbias[n0 + l16];
    #pragma unroll
    for (int sub = 0; sub < 4; ++sub) {
        #pragma unroll
        for (int v = 0; v < 8; ++v) {
            int m = m0 + sub*16 + hi*8 + v;   // C/D layout: M = v + 8*hi
            float val = acc[sub][v] + bias;
            val = val > 0.f ? val : 0.f;
            H[(size_t)m * NOUT + n0 + l16] = (_Float16)val;
        }
    }
}

// ---------------------------------------------------------------------------
// Kernel 6: fc2 [8192,512]x[512,10] + tanh. One thread per output element.
// ---------------------------------------------------------------------------
__global__ __launch_bounds__(256) void fc2_kernel(
    const _Float16* __restrict__ H, const float* __restrict__ w2,
    const float* __restrict__ b2, float* __restrict__ out)
{
    int idx = blockIdx.x * 256 + threadIdx.x;
    if (idx >= 8192 * 10) return;
    int r = idx / 10, j = idx % 10;
    const v8h* hp = (const v8h*)(H + (size_t)r * NOUT);
    float acc = b2[j];
    for (int t = 0; t < 64; ++t) {
        v8h hv = hp[t];
        #pragma unroll
        for (int u = 0; u < 8; ++u) acc += (float)hv[u] * w2[(t*8 + u)*10 + j];
    }
    out[idx] = tanhf(acc);
}

// ---------------------------------------------------------------------------
extern "C" void kernel_launch(void* const* d_in, const int* in_sizes, int n_in,
                              void* d_out, int out_size, void* d_ws, size_t ws_size,
                              hipStream_t stream)
{
    (void)in_sizes; (void)n_in; (void)out_size; (void)ws_size;
    const float* node_fea = (const float*)d_in[0];
    const float* edge_fea = (const float*)d_in[1];
    const float* pic      = (const float*)d_in[2];
    const float* c1w = (const float*)d_in[3];  const float* c1b = (const float*)d_in[4];
    const float* c2w = (const float*)d_in[5];  const float* c2b = (const float*)d_in[6];
    const float* g1w = (const float*)d_in[7];  const float* g1as = (const float*)d_in[8];
    const float* g1ad = (const float*)d_in[9]; const float* g1le = (const float*)d_in[10];
    const float* g1ae = (const float*)d_in[11];const float* g1b = (const float*)d_in[12];
    const float* g2w = (const float*)d_in[13]; const float* g2as = (const float*)d_in[14];
    const float* g2ad = (const float*)d_in[15];const float* g2le = (const float*)d_in[16];
    const float* g2ae = (const float*)d_in[17];const float* g2b = (const float*)d_in[18];
    const float* fc1w = (const float*)d_in[19];const float* fc1b = (const float*)d_in[20];
    const float* fc2w = (const float*)d_in[21];const float* fc2b = (const float*)d_in[22];
    const int* eidx = (const int*)d_in[23];
    float* out = (float*)d_out;

    // workspace layout (all regions fully overwritten every call)
    char* w = (char*)d_ws;
    float*    gbias = (float*)w;                                   //      2,048 B
    float*    p1    = (float*)(w + 2048);                          // 38,535,168 B
    _Float16* A     = (_Float16*)(w + 2048 + 38535168ULL);         //  6,815,744 B
    _Float16* Bt    = A + (size_t)8192 * KP;                       //    425,984 B
    _Float16* H     = Bt + (size_t)NOUT * KP;                      //  8,388,608 B

    gnn_kernel<<<1, 256, 0, stream>>>(node_fea, edge_fea, eidx,
        g1w, g1as, g1ad, g1le, g1ae, g1b,
        g2w, g2as, g2ad, g2le, g2ae, g2b,
        fc1w, fc1b, gbias);
    conv1_kernel<<<37632, 256, 0, stream>>>(pic, c1w, c1b, p1);
    conv2_kernel<<<13312, 256, 0, stream>>>(p1, c2w, c2b, A);
    prepb_kernel<<<832, 256, 0, stream>>>(fc1w, Bt);
    fc1_wmma_kernel<<<512, 256, 0, stream>>>(A, Bt, gbias, H);
    fc2_kernel<<<320, 256, 0, stream>>>(H, fc2w, fc2b, out);
}
